// Multiheadattention_48112223650540
// MI455X (gfx1250) — matrix operand order repaired and checked by reading, not compiled, not run on hardware
//
#include <hip/hip_runtime.h>
#include <hip/hip_bf16.h>
#include <math.h>
#include <stdint.h>

#define EMBED 1024
#define HEADS 16
#define HSIZE 64
#define SEQ   2048

typedef __attribute__((ext_vector_type(16))) _Float16 v16h;
typedef __attribute__((ext_vector_type(8)))  float    v8f;

union Frag16 { v16h h; uint4 u[2]; _Float16 e[16]; };

// ---------------- CDNA5 async global->LDS path (guarded; falls back to sync copy)
#if defined(__has_builtin)
#if __has_builtin(__builtin_amdgcn_global_load_async_to_lds_b128)
#define ASYNC_LDS 1
#endif
#endif

// builtin parameter types: pointer-to-int4 (gcc vector_size(16)) in AS1 / AS3
typedef int i32x4 __attribute__((__vector_size__(16)));
typedef __attribute__((address_space(1))) i32x4 gi32x4;
typedef __attribute__((address_space(3))) i32x4 li32x4;

// Copy 16 bytes global -> LDS. Async variant lowers to global_load_async_to_lds_b128
// (per-lane: each lane copies 16B from its global address to its LDS address, tracked by ASYNCcnt; no VGPR round trip).
__device__ __forceinline__ void cp16(_Float16* dstL, const _Float16* srcG) {
#ifdef ASYNC_LDS
  __builtin_amdgcn_global_load_async_to_lds_b128(
      (gi32x4*)(uintptr_t)srcG,
      (li32x4*)(unsigned int)(uintptr_t)dstL, 0, 0);
#else
  *(uint4*)dstL = *(const uint4*)srcG;
#endif
}

__device__ __forceinline__ void wait_async() {
#ifdef ASYNC_LDS
#if __has_builtin(__builtin_amdgcn_s_wait_asynccnt)
  __builtin_amdgcn_s_wait_asynccnt(0);
#else
  asm volatile("s_wait_asynccnt 0x0" ::: "memory");
#endif
#endif
}

// A-operand fragment (16x32 f16): lane = M row; lanes 0-15 hold K {0..7,16..23},
// lanes 16-31 hold K {8..15,24..31}  -> two 16B chunks at kb and kb+16 halfs.
__device__ __forceinline__ v16h ld_fragA(const _Float16* p, int stride) {
  int lane = threadIdx.x & 31;
  int row  = lane & 15;
  int kb   = (lane >> 4) << 3;   // 0 or 8
  Frag16 f;
  f.u[0] = *(const uint4*)(p + (size_t)row * stride + kb);
  f.u[1] = *(const uint4*)(p + (size_t)row * stride + kb + 16);
  return f.h;
}

// B-operand fragment (32x16 f16), sourced from rows of B^T (row-major, K contiguous):
// lane = N col; lanes 0-15 hold K {0..7,16..23}, lanes 16-31 hold K {8..15,24..31}.
__device__ __forceinline__ v16h ld_fragB(const _Float16* p, int stride) {
  int lane = threadIdx.x & 31;
  int col  = lane & 15;
  int kb   = (lane >> 4) << 3;   // 0 or 8
  Frag16 f;
  f.u[0] = *(const uint4*)(p + (size_t)col * stride + kb);
  f.u[1] = *(const uint4*)(p + (size_t)col * stride + kb + 16);
  return f.h;
}

#define WMMA_F16(a, b, c) \
  __builtin_amdgcn_wmma_f32_16x16x32_f16(false, (a), false, (b), (short)0, (c), false, false)

// ---------------------------------------------------------------- converts
__global__ void cvt_f32_f16(const float* __restrict__ s, _Float16* __restrict__ d, int n) {
  int i = (blockIdx.x * blockDim.x + threadIdx.x) * 4;
  int stride = gridDim.x * blockDim.x * 4;
  for (; i < n; i += stride) {
    float4 v = *(const float4*)(s + i);
    _Float16 o[4] = {(_Float16)v.x, (_Float16)v.y, (_Float16)v.z, (_Float16)v.w};
    *(uint2*)(d + i) = *(const uint2*)o;
  }
}

// ---------------------------------------------------------------- GEMM: C = A * W^T + bias
// A [M,K] f16 row-major, W [N,K] f16 row-major (torch Linear weight), bias [N] f32.
// Block tile 128x128, K-step 32, 256 threads = 8 waves in a 2x4 wave grid,
// each wave: 64x32 = 4x2 fragments. Double-buffered LDS, async global->LDS staging.
// LDS row stride 40 halfs => conflict-free ds_load_b128 fragment fetches.
#define GSTR 40
__global__ __launch_bounds__(256) void gemm_xwT(
    const _Float16* __restrict__ A, const _Float16* __restrict__ W,
    const float* __restrict__ bias,
    _Float16* __restrict__ outH, float* __restrict__ outF,
    int M, int N, int K)
{
  __shared__ _Float16 sA[2][128 * GSTR];
  __shared__ _Float16 sB[2][128 * GSTR];
  const int tid  = threadIdx.x;
  const int lane = tid & 31;
  const int w    = tid >> 5;
  const int wm   = w >> 2;      // 0..1
  const int wn   = w & 3;       // 0..3
  const int m0   = blockIdx.y * 128;
  const int n0   = blockIdx.x * 128;

  // staging coords: 128 rows x 4 chunks(8 halfs) per array; each thread: 2 rows (r, r+64)
  const int srow = tid >> 2;
  const int scc  = (tid & 3) << 3;
  const _Float16* gA = A + (size_t)(m0 + srow) * K + scc;
  const _Float16* gB = W + (size_t)(n0 + srow) * K + scc;
  const size_t rstep = (size_t)64 * K;
  const int l0 = srow * GSTR + scc;
  const int l1 = l0 + 64 * GSTR;

  auto stage = [&](int buf, int k0) {
    cp16(&sA[buf][l0], gA + k0);
    cp16(&sA[buf][l1], gA + rstep + k0);
    cp16(&sB[buf][l0], gB + k0);
    cp16(&sB[buf][l1], gB + rstep + k0);
  };

  v8f acc[4][2];
#pragma unroll
  for (int mt = 0; mt < 4; ++mt)
#pragma unroll
    for (int nt = 0; nt < 2; ++nt)
#pragma unroll
      for (int i = 0; i < 8; ++i) acc[mt][nt][i] = 0.0f;

  stage(0, 0);
  int buf = 0;
  for (int k0 = 0; k0 < K; k0 += 32) {
    wait_async();        // this wave's pending tile copies done
    __syncthreads();     // everyone's copies visible; prior reads of other buffer done
    if (k0 + 32 < K) stage(buf ^ 1, k0 + 32);   // prefetch next tile (other buffer)

    v16h af[4], bf[2];
#pragma unroll
    for (int mt = 0; mt < 4; ++mt) af[mt] = ld_fragA(&sA[buf][(wm * 64 + mt * 16) * GSTR], GSTR);
#pragma unroll
    for (int nt = 0; nt < 2; ++nt) bf[nt] = ld_fragB(&sB[buf][(wn * 32 + nt * 16) * GSTR], GSTR);
#pragma unroll
    for (int mt = 0; mt < 4; ++mt)
#pragma unroll
      for (int nt = 0; nt < 2; ++nt)
        acc[mt][nt] = WMMA_F16(af[mt], bf[nt], acc[mt][nt]);
    buf ^= 1;
  }

  const int half = lane >> 4, lx = lane & 15;
#pragma unroll
  for (int mt = 0; mt < 4; ++mt)
#pragma unroll
    for (int nt = 0; nt < 2; ++nt) {
      int n = n0 + wn * 32 + nt * 16 + lx;
      float bn = bias[n];
#pragma unroll
      for (int r = 0; r < 8; ++r) {
        int m = m0 + wm * 64 + mt * 16 + r + 8 * half;
        float v = acc[mt][nt][r] + bn;
        if (outF) outF[(size_t)m * N + n] = v;
        else      outH[(size_t)m * N + n] = (_Float16)v;
      }
    }
}

// ---------------------------------------------------------------- fused attention
// grid = (SEQ/128, B*HEADS); 8 waves/block, each wave owns 16 query rows.
// Flash-style online softmax over 64-wide K/V tiles staged in LDS
// (K tile via async global->LDS, V tile transposed via sync scatter).
#define ASTR 72
__global__ __launch_bounds__(256) void attn_kernel(
    const _Float16* __restrict__ Q, const _Float16* __restrict__ Km,
    const _Float16* __restrict__ V, _Float16* __restrict__ O,
    int S, float invscale)
{
  __shared__ _Float16 sK[64 * ASTR];        // K tile rows (d contiguous)
  __shared__ _Float16 sV[64 * ASTR];        // V^T tile: sV[d*ASTR + kk]
  __shared__ _Float16 sP[8 * 16 * ASTR];    // per-wave P scratch (C-layout -> A-layout)

  const int tid  = threadIdx.x;
  const int lane = tid & 31;
  const int w    = tid >> 5;
  const int half = lane >> 4;
  const int lx   = lane & 15;
  const int b    = blockIdx.y >> 4;
  const int h    = blockIdx.y & 15;
  const int q0   = blockIdx.x * 128 + w * 16;

  const size_t baseQ = ((size_t)b * S + q0) * EMBED + h * HSIZE;
  v16h qa0 = ld_fragA(Q + baseQ, EMBED);        // d 0..31
  v16h qa1 = ld_fragA(Q + baseQ + 32, EMBED);   // d 32..63

  // staging coords: K tile = 64 rows x 8 chunks; thread covers rows krow, krow+32
  const int krow = tid >> 3;
  const int kcc  = (tid & 7) << 3;
  const int vrow = tid >> 2;            // V: 4 threads/row, 16 halfs each
  const int vc0  = (tid & 3) << 4;

  float mrun[8], lrun[8];
  v8f oacc[4];
#pragma unroll
  for (int r = 0; r < 8; ++r) { mrun[r] = -1e30f; lrun[r] = 0.0f; }
#pragma unroll
  for (int dt = 0; dt < 4; ++dt)
#pragma unroll
    for (int i = 0; i < 8; ++i) oacc[dt][i] = 0.0f;

  _Float16* sPw = &sP[w * 16 * ASTR];

  for (int j = 0; j < S / 64; ++j) {
    __syncthreads();   // previous iteration's reads of sK/sV complete
    // K tile: async copies straight into LDS
    {
      const _Float16* src = &Km[((size_t)b * S + j * 64 + krow) * EMBED + h * HSIZE + kcc];
      cp16(&sK[krow * ASTR + kcc], src);
      cp16(&sK[(krow + 32) * ASTR + kcc], src + (size_t)32 * EMBED);
    }
    // V tile transposed (scatter by d), overlaps with K async copies in flight
    {
      Frag16 f;
      const _Float16* src = &V[((size_t)b * S + j * 64 + vrow) * EMBED + h * HSIZE + vc0];
      f.u[0] = *(const uint4*)src;
      f.u[1] = *(const uint4*)(src + 8);
#pragma unroll
      for (int i = 0; i < 16; ++i) sV[(vc0 + i) * ASTR + vrow] = f.e[i];
    }
    wait_async();
    __syncthreads();

    // scores: S = Q (16x64) . K^T (64x64)  -> four 16x16 C tiles
    v8f sc[4];
#pragma unroll
    for (int nt = 0; nt < 4; ++nt) {
      v8f z;
#pragma unroll
      for (int i = 0; i < 8; ++i) z[i] = 0.0f;
      v16h kb0 = ld_fragB(&sK[nt * 16 * ASTR], ASTR);
      v16h kb1 = ld_fragB(&sK[nt * 16 * ASTR + 32], ASTR);
      z = WMMA_F16(qa0, kb0, z);
      z = WMMA_F16(qa1, kb1, z);
#pragma unroll
      for (int i = 0; i < 8; ++i) z[i] *= invscale;
      sc[nt] = z;
    }

    // online softmax per row (row r lives in lanes of one 16-lane half)
#pragma unroll
    for (int r = 0; r < 8; ++r) {
      float xm = sc[0][r];
#pragma unroll
      for (int nt = 1; nt < 4; ++nt) xm = fmaxf(xm, sc[nt][r]);
#pragma unroll
      for (int off = 1; off < 16; off <<= 1) xm = fmaxf(xm, __shfl_xor(xm, off, 32));
      float mnew = fmaxf(mrun[r], xm);
      float corr = __expf(mrun[r] - mnew);
      float ps = 0.0f;
#pragma unroll
      for (int nt = 0; nt < 4; ++nt) {
        float p = __expf(sc[nt][r] - mnew);
        ps += p;
        sPw[(r + 8 * half) * ASTR + nt * 16 + lx] = (_Float16)p;  // C-layout store
      }
#pragma unroll
      for (int off = 1; off < 16; off <<= 1) ps += __shfl_xor(ps, off, 32);
      lrun[r] = lrun[r] * corr + ps;
      mrun[r] = mnew;
#pragma unroll
      for (int dt = 0; dt < 4; ++dt) oacc[dt][r] *= corr;
    }

    // reload P in A-layout (per-wave scratch; in-wave LDS ordering suffices)
    v16h pa0 = ld_fragA(sPw, ASTR);        // keys 0..31
    v16h pa1 = ld_fragA(sPw + 32, ASTR);   // keys 32..63
#pragma unroll
    for (int dt = 0; dt < 4; ++dt) {
      v16h vb0 = ld_fragB(&sV[dt * 16 * ASTR], ASTR);
      v16h vb1 = ld_fragB(&sV[dt * 16 * ASTR + 32], ASTR);
      oacc[dt] = WMMA_F16(pa0, vb0, oacc[dt]);
      oacc[dt] = WMMA_F16(pa1, vb1, oacc[dt]);
    }
  }

  // epilogue: divide by l, write [B,S,E] with head offset
#pragma unroll
  for (int dt = 0; dt < 4; ++dt)
#pragma unroll
    for (int r = 0; r < 8; ++r) {
      int mrow = q0 + r + 8 * half;
      int col  = h * HSIZE + dt * 16 + lx;
      float o = oacc[dt][r] / lrun[r];
      O[((size_t)b * S + mrow) * EMBED + col] = (_Float16)o;
    }
}

// ---------------------------------------------------------------- host
extern "C" void kernel_launch(void* const* d_in, const int* in_sizes, int n_in,
                              void* d_out, int out_size, void* d_ws, size_t ws_size,
                              hipStream_t stream) {
  const float* querys = (const float*)d_in[0];
  const float* keys   = (const float*)d_in[1];
  const float* values = (const float*)d_in[2];
  const float* Wq = (const float*)d_in[3];  const float* bq = (const float*)d_in[4];
  const float* Wk = (const float*)d_in[5];  const float* bk = (const float*)d_in[6];
  const float* Wv = (const float*)d_in[7];  const float* bv = (const float*)d_in[8];
  const float* Wo = (const float*)d_in[9];  const float* bo = (const float*)d_in[10];

  const int BS_E = in_sizes[0];          // B*S*E
  const int Mrows = BS_E / EMBED;        // B*S = 4096
  const int Bsz = Mrows / SEQ;           // 2

  _Float16* ws = (_Float16*)d_ws;
  const size_t szIn = (size_t)Mrows * EMBED;
  const size_t szW  = (size_t)EMBED * EMBED;
  _Float16* qh  = ws;
  _Float16* kh  = qh  + szIn;
  _Float16* vh  = kh  + szIn;
  _Float16* wqh = vh  + szIn;
  _Float16* wkh = wqh + szW;
  _Float16* wvh = wkh + szW;
  _Float16* woh = wvh + szW;
  _Float16* Qp  = woh + szW;
  _Float16* Kp  = Qp  + szIn;
  _Float16* Vp  = Kp  + szIn;
  _Float16* Ao  = Vp  + szIn;

  // fp32 -> fp16 converts
  cvt_f32_f16<<<1024, 256, 0, stream>>>(querys, qh, BS_E);
  cvt_f32_f16<<<1024, 256, 0, stream>>>(keys,   kh, BS_E);
  cvt_f32_f16<<<1024, 256, 0, stream>>>(values, vh, BS_E);
  cvt_f32_f16<<<512,  256, 0, stream>>>(Wq, wqh, (int)szW);
  cvt_f32_f16<<<512,  256, 0, stream>>>(Wk, wkh, (int)szW);
  cvt_f32_f16<<<512,  256, 0, stream>>>(Wv, wvh, (int)szW);
  cvt_f32_f16<<<512,  256, 0, stream>>>(Wo, woh, (int)szW);

  // projections
  dim3 gblk(256), ggrid(EMBED / 128, Mrows / 128);
  gemm_xwT<<<ggrid, gblk, 0, stream>>>(qh, wqh, bq, Qp, nullptr, Mrows, EMBED, EMBED);
  gemm_xwT<<<ggrid, gblk, 0, stream>>>(kh, wkh, bk, Kp, nullptr, Mrows, EMBED, EMBED);
  gemm_xwT<<<ggrid, gblk, 0, stream>>>(vh, wvh, bv, Vp, nullptr, Mrows, EMBED, EMBED);

  // fused flash attention (note: reference scale is D**0.05, faithful to source)
  float invscale = 1.0f / powf((float)HSIZE, 0.05f);
  dim3 agrid(SEQ / 128, Bsz * HEADS);
  attn_kernel<<<agrid, 256, 0, stream>>>(Qp, Kp, Vp, Ao, SEQ, invscale);

  // output projection, f32 out + bias
  gemm_xwT<<<ggrid, gblk, 0, stream>>>(Ao, woh, bo, nullptr, (float*)d_out, Mrows, EMBED, EMBED);
}
